// HierDSFeedForward_71451075936561
// MI455X (gfx1250) — matrix operand order repaired and checked by reading, not compiled, not run on hardware
//
#include <hip/hip_runtime.h>
#include <hip/hip_bf16.h>

typedef __bf16 bf16_t;
typedef __attribute__((ext_vector_type(16))) __bf16 v16bf;
typedef __attribute__((ext_vector_type(8)))  __bf16 v8bf;
typedef __attribute__((ext_vector_type(8)))  float  v8f;

// global-address-space views (force global_load lowering, avoid flat path)
typedef __attribute__((address_space(1))) const __bf16 gbf_t;
typedef __attribute__((address_space(1))) const v8bf  gv8bf_t;

#define S_TOK  2048   // B*T
#define DMODEL 1024
#define HDIM   2048
#define NEXP   8
#define EPG    4
#define CAP    2048   // per-expert token capacity
#define LN_EPS 1e-5f

// ---------------------------------------------------------------- helpers

__device__ __forceinline__ v8f zero8() {
    v8f z;
#pragma unroll
    for (int i = 0; i < 8; ++i) z[i] = 0.f;
    return z;
}

__device__ __forceinline__ v8f wmma_bf16(v16bf a, v16bf b, v8f c) {
    return __builtin_amdgcn_wmma_f32_16x16x32_bf16(
        false, a, false, b, (short)0, c, false, false);
}

// A fragment (16x32, 16-bit): lane<16 holds row m, K = k0+0..7 and k0+16..23;
// lane>=16 holds row m, K = k0+8..15 and k0+24..31.
// p must point at (row0+m)*lda + k0 + half*8.
__device__ __forceinline__ v16bf frag_a_at(gbf_t* p) {
    union { v16bf v; struct { v8bf lo, hi; } s; } u;
    u.s.lo = *(gv8bf_t*)(p);
    u.s.hi = *(gv8bf_t*)(p + 16);
    return u.v;
}

// B fragment (32x16, 16-bit) from B^T row-major [N, K]:
// lane<16 holds column n, K = k0+0..7 and k0+16..23; lane>=16: K = k0+8..15 and k0+24..31.
// p must point at (col0+n)*ldb + k0 + half*8.
__device__ __forceinline__ v16bf frag_b_at(gbf_t* p) {
    union { v16bf v; struct { v8bf lo, hi; } s; } u;
    u.s.lo = *(gv8bf_t*)(p);
    u.s.hi = *(gv8bf_t*)(p + 16);
    return u.v;
}

__device__ __forceinline__ gbf_t* a_base(gbf_t* base, int lda, int row0, int lane) {
    return base + (size_t)(row0 + (lane & 15)) * lda + (lane >> 4) * 8;
}
__device__ __forceinline__ gbf_t* b_base(gbf_t* bt, int ldb, int col0, int lane) {
    return bt + (size_t)(col0 + (lane & 15)) * ldb + (lane >> 4) * 8;
}

// ---------------------------------------------------------------- layernorm

__global__ void ln_kernel(const float* __restrict__ x,
                          const float* __restrict__ sc,
                          const float* __restrict__ bi,
                          bf16_t* __restrict__ out) {
    int wave = (blockIdx.x * blockDim.x + threadIdx.x) >> 5;
    int lane = threadIdx.x & 31;
    if (wave >= S_TOK) return;
    const float4* xr = (const float4*)(x + (size_t)wave * DMODEL);
    float4 v[8];
    float s = 0.f, ss = 0.f;
#pragma unroll
    for (int i = 0; i < 8; ++i) {
        v[i] = xr[i * 32 + lane];
        s  += v[i].x + v[i].y + v[i].z + v[i].w;
        ss += v[i].x * v[i].x + v[i].y * v[i].y + v[i].z * v[i].z + v[i].w * v[i].w;
    }
#pragma unroll
    for (int off = 16; off > 0; off >>= 1) {
        s  += __shfl_xor(s,  off, 32);
        ss += __shfl_xor(ss, off, 32);
    }
    float mean = s * (1.f / DMODEL);
    float var  = ss * (1.f / DMODEL) - mean * mean;
    float rstd = rsqrtf(var + LN_EPS);
    bf16_t* orow = out + (size_t)wave * DMODEL;
#pragma unroll
    for (int i = 0; i < 8; ++i) {
        int d = (i * 32 + lane) * 4;
        float e0 = (v[i].x - mean) * rstd * sc[d + 0] + bi[d + 0];
        float e1 = (v[i].y - mean) * rstd * sc[d + 1] + bi[d + 1];
        float e2 = (v[i].z - mean) * rstd * sc[d + 2] + bi[d + 2];
        float e3 = (v[i].w - mean) * rstd * sc[d + 3] + bi[d + 3];
        orow[d + 0] = (bf16_t)e0; orow[d + 1] = (bf16_t)e1;
        orow[d + 2] = (bf16_t)e2; orow[d + 3] = (bf16_t)e3;
    }
}

// -------------------------------------------------- weight transpose/convert

// out[c*R + r] = (bf16) in[r*C + c]; R, C multiples of 32.
__global__ void transpose_cvt_kernel(const float* __restrict__ in,
                                     bf16_t* __restrict__ out, int R, int C) {
    __shared__ float tile[32][33];
    int c0 = blockIdx.x * 32, r0 = blockIdx.y * 32;
    int tx = threadIdx.x, ty = threadIdx.y;   // 32 x 8
#pragma unroll
    for (int i = 0; i < 32; i += 8)
        tile[ty + i][tx] = in[(size_t)(r0 + ty + i) * C + (c0 + tx)];
    __syncthreads();
#pragma unroll
    for (int i = 0; i < 32; i += 8)
        out[(size_t)(c0 + ty + i) * R + (r0 + tx)] = (bf16_t)tile[tx][ty + i];
}

__global__ void cvt_kernel(const float* __restrict__ in, bf16_t* __restrict__ out, int n) {
    int i = blockIdx.x * blockDim.x + threadIdx.x;
    if (i < n) out[i] = (bf16_t)in[i];
}

// ---------------------------------------------------------------- routing

__global__ void init_kernel(int* __restrict__ stok, float* __restrict__ sw,
                            int* __restrict__ sslot, int* __restrict__ cursor,
                            bf16_t* __restrict__ hx) {
    int i = blockIdx.x * blockDim.x + threadIdx.x;      // 16384 threads
    if (i < NEXP * CAP) { stok[i] = S_TOK; sw[i] = 0.f; sslot[i] = 0; }
    if (i < NEXP) cursor[i] = 0;
    if (i < HDIM) hx[(size_t)S_TOK * HDIM + i] = (bf16_t)0.f;   // dummy row
}

__global__ void route_kernel(const bf16_t* __restrict__ flat,
                             const float* __restrict__ wg,
                             const float* __restrict__ we,
                             const float* __restrict__ gb,
                             const float* __restrict__ eb,
                             int* __restrict__ stok, float* __restrict__ sw,
                             int* __restrict__ sslot, int* __restrict__ cursor) {
    int lane  = threadIdx.x & 31;
    int token = blockIdx.x * (blockDim.x >> 5) + (threadIdx.x >> 5);
    if (token >= S_TOK) return;
    float ag0 = 0.f, ag1 = 0.f;
    float ae[NEXP];
#pragma unroll
    for (int e = 0; e < NEXP; ++e) ae[e] = 0.f;
    const bf16_t* arow = flat + (size_t)token * DMODEL;
    for (int d = lane; d < DMODEL; d += 32) {
        float a = (float)arow[d];
        ag0 += a * wg[d * 2 + 0];
        ag1 += a * wg[d * 2 + 1];
#pragma unroll
        for (int e = 0; e < NEXP; ++e) ae[e] += a * we[d * NEXP + e];
    }
#pragma unroll
    for (int off = 16; off > 0; off >>= 1) {
        ag0 += __shfl_xor(ag0, off, 32);
        ag1 += __shfl_xor(ag1, off, 32);
#pragma unroll
        for (int e = 0; e < NEXP; ++e) ae[e] += __shfl_xor(ae[e], off, 32);
    }
    if (lane == 0) {
        float g0 = ag0 + gb[0], g1 = ag1 + gb[1];
        int g = (g1 > g0) ? 1 : 0;
        float p[EPG];
        float mx = -1e30f;
#pragma unroll
        for (int j = 0; j < EPG; ++j) {
            p[j] = ae[g * EPG + j] + eb[g * EPG + j];
            mx = fmaxf(mx, p[j]);
        }
        float sum = 0.f;
#pragma unroll
        for (int j = 0; j < EPG; ++j) { p[j] = __expf(p[j] - mx); sum += p[j]; }
        float inv = 1.f / sum;
#pragma unroll
        for (int j = 0; j < EPG; ++j) p[j] *= inv;
        int i0 = 0; float v0 = p[0];
#pragma unroll
        for (int j = 1; j < EPG; ++j) if (p[j] > v0) { v0 = p[j]; i0 = j; }
        int i1 = -1; float v1 = -1e30f;
#pragma unroll
        for (int j = 0; j < EPG; ++j) if (j != i0 && p[j] > v1) { v1 = p[j]; i1 = j; }
        int e0 = g * EPG + i0, e1 = g * EPG + i1;
        int pos0 = atomicAdd(&cursor[e0], 1);
        stok[e0 * CAP + pos0] = token; sw[e0 * CAP + pos0] = v0; sslot[e0 * CAP + pos0] = 0;
        int pos1 = atomicAdd(&cursor[e1], 1);
        stok[e1 * CAP + pos1] = token; sw[e1 * CAP + pos1] = v1; sslot[e1 * CAP + pos1] = 1;
    }
}

// ------------------------------------------- GEMM 1: in-proj + fused SwiGLU
// A: flat bf16 [S, DMODEL]; Bt: W^T bf16 [2H, DMODEL]; out: h bf16 [S, HDIM]
// wave computes 32 rows x 64 cols of both the a-half and b-half
// (each B fragment feeds two WMMAs). Pointers advance 32 elems per K-step.

__global__ void __launch_bounds__(256, 1)
gemm_swiglu_kernel(const bf16_t* __restrict__ A,
                   const bf16_t* __restrict__ Bt,
                   bf16_t* __restrict__ Hout) {
    int lane = threadIdx.x & 31;
    int wave = threadIdx.x >> 5;
    int row0 = blockIdx.x * 256 + wave * 32;
    int c0   = blockIdx.y * 64;
    gbf_t* Ag  = (gbf_t*)A;
    gbf_t* Btg = (gbf_t*)Bt;
    v8f aa[2][4], ab[2][4];
#pragma unroll
    for (int rs = 0; rs < 2; ++rs)
#pragma unroll
        for (int j = 0; j < 4; ++j) { aa[rs][j] = zero8(); ab[rs][j] = zero8(); }

    gbf_t* pa0 = a_base(Ag, DMODEL, row0,      lane);
    gbf_t* pa1 = a_base(Ag, DMODEL, row0 + 16, lane);
    gbf_t* pb[8];
#pragma unroll
    for (int j = 0; j < 4; ++j) {
        pb[j]     = b_base(Btg, DMODEL, c0 + 16 * j,        lane);
        pb[j + 4] = b_base(Btg, DMODEL, HDIM + c0 + 16 * j, lane);
    }
#pragma unroll 1
    for (int k0 = 0; k0 < DMODEL; k0 += 32) {
        v16bf af0 = frag_a_at(pa0);
        v16bf af1 = frag_a_at(pa1);
#pragma unroll
        for (int j = 0; j < 4; ++j) {
            v16bf b0 = frag_b_at(pb[j]);
            aa[0][j] = wmma_bf16(af0, b0, aa[0][j]);
            aa[1][j] = wmma_bf16(af1, b0, aa[1][j]);
            v16bf b1 = frag_b_at(pb[j + 4]);
            ab[0][j] = wmma_bf16(af0, b1, ab[0][j]);
            ab[1][j] = wmma_bf16(af1, b1, ab[1][j]);
        }
        pa0 += 32; pa1 += 32;
#pragma unroll
        for (int j = 0; j < 8; ++j) pb[j] += 32;
    }
    int n = lane & 15, half = lane >> 4;
#pragma unroll
    for (int rs = 0; rs < 2; ++rs) {
#pragma unroll
        for (int j = 0; j < 4; ++j) {
            int col = c0 + 16 * j + n;
#pragma unroll
            for (int r = 0; r < 8; ++r) {
                float a = aa[rs][j][r];
                float b = ab[rs][j][r];
                float h = (a / (1.f + __expf(-a))) * b;   // silu(a) * b
                Hout[(size_t)(row0 + rs * 16 + r + half * 8) * HDIM + col] = (bf16_t)h;
            }
        }
    }
}

// ------------------------------------------- GEMM 2: shared out-proj (+bias)
// A: h_shared bf16 [S, HDIM]; Bt: Wso^T bf16 [DMODEL, HDIM]; out fp32 [S, DMODEL]

__global__ void __launch_bounds__(256, 1)
gemm_bias_kernel(const bf16_t* __restrict__ A,
                 const bf16_t* __restrict__ Bt,
                 const float* __restrict__ bias,
                 float* __restrict__ Out) {
    int lane = threadIdx.x & 31;
    int wave = threadIdx.x >> 5;
    int row0 = blockIdx.x * 256 + wave * 32;
    int c0   = blockIdx.y * 64;
    gbf_t* Ag  = (gbf_t*)A;
    gbf_t* Btg = (gbf_t*)Bt;
    v8f acc[2][4];
#pragma unroll
    for (int rs = 0; rs < 2; ++rs)
#pragma unroll
        for (int j = 0; j < 4; ++j) acc[rs][j] = zero8();

    gbf_t* pa0 = a_base(Ag, HDIM, row0,      lane);
    gbf_t* pa1 = a_base(Ag, HDIM, row0 + 16, lane);
    gbf_t* pb[4];
#pragma unroll
    for (int j = 0; j < 4; ++j) pb[j] = b_base(Btg, HDIM, c0 + 16 * j, lane);

#pragma unroll 1
    for (int k0 = 0; k0 < HDIM; k0 += 32) {
        v16bf af0 = frag_a_at(pa0);
        v16bf af1 = frag_a_at(pa1);
#pragma unroll
        for (int j = 0; j < 4; ++j) {
            v16bf bf = frag_b_at(pb[j]);
            acc[0][j] = wmma_bf16(af0, bf, acc[0][j]);
            acc[1][j] = wmma_bf16(af1, bf, acc[1][j]);
        }
        pa0 += 32; pa1 += 32;
#pragma unroll
        for (int j = 0; j < 4; ++j) pb[j] += 32;
    }
    int n = lane & 15, half = lane >> 4;
#pragma unroll
    for (int rs = 0; rs < 2; ++rs) {
#pragma unroll
        for (int j = 0; j < 4; ++j) {
            int col = c0 + 16 * j + n;
            float bv = bias[col];
#pragma unroll
            for (int r = 0; r < 8; ++r)
                Out[(size_t)(row0 + rs * 16 + r + half * 8) * DMODEL + col] =
                    acc[rs][j][r] + bv;
        }
    }
}

// --------------------------------- GEMM 3: grouped expert out-proj (gather)
// A: h_expert bf16 [S+16, HDIM] gathered via sorted token list;
// B: expert_out_weight bf16 [E, DMODEL, HDIM] (already B^T layout);
// writes w*(acc + expert_bias) into per-slot buffers rk0/rk1 [S+16, DMODEL].
// One wave per block; 32 sorted tokens x 64 cols.

__global__ void __launch_bounds__(32, 1)
gemm_expert_kernel(const bf16_t* __restrict__ Hx,
                   const bf16_t* __restrict__ Weo,
                   const float* __restrict__ EOB,
                   const int* __restrict__ stok,
                   const float* __restrict__ sw,
                   const int* __restrict__ sslot,
                   const int* __restrict__ cursor,
                   float* __restrict__ rk0,
                   float* __restrict__ rk1) {
    int e   = blockIdx.z;
    int t0  = blockIdx.x * 32;
    if (t0 >= cursor[e]) return;                        // past live tokens
    int c0  = blockIdx.y * 64;
    int lane = threadIdx.x;
    int m = lane & 15, half = lane >> 4;
    int seg = e * CAP;
    int tok0 = stok[seg + t0 + m];                      // per-lane gather rows
    int tok1 = stok[seg + t0 + 16 + m];
    gbf_t* Hxg = (gbf_t*)Hx;
    gbf_t* Btg = (gbf_t*)Weo + (size_t)e * DMODEL * HDIM;   // [DMODEL, HDIM]
    v8f acc[2][4];
#pragma unroll
    for (int rs = 0; rs < 2; ++rs)
#pragma unroll
        for (int j = 0; j < 4; ++j) acc[rs][j] = zero8();

    gbf_t* pa0 = Hxg + (size_t)tok0 * HDIM + half * 8;
    gbf_t* pa1 = Hxg + (size_t)tok1 * HDIM + half * 8;
    gbf_t* pb[4];
#pragma unroll
    for (int j = 0; j < 4; ++j) pb[j] = b_base(Btg, HDIM, c0 + 16 * j, lane);

#pragma unroll 1
    for (int k0 = 0; k0 < HDIM; k0 += 32) {
        v16bf af0 = frag_a_at(pa0);
        v16bf af1 = frag_a_at(pa1);
#pragma unroll
        for (int j = 0; j < 4; ++j) {
            v16bf bf = frag_b_at(pb[j]);
            acc[0][j] = wmma_bf16(af0, bf, acc[0][j]);
            acc[1][j] = wmma_bf16(af1, bf, acc[1][j]);
        }
        pa0 += 32; pa1 += 32;
#pragma unroll
        for (int j = 0; j < 4; ++j) pb[j] += 32;
    }
#pragma unroll
    for (int rs = 0; rs < 2; ++rs) {
#pragma unroll
        for (int j = 0; j < 4; ++j) {
            int col = c0 + 16 * j + m;
            float be = EOB[(size_t)e * DMODEL + col];
#pragma unroll
            for (int r = 0; r < 8; ++r) {
                int idx  = seg + t0 + rs * 16 + r + half * 8;
                int tok  = stok[idx];
                float w  = sw[idx];
                float* dst = sslot[idx] ? rk1 : rk0;
                dst[(size_t)tok * DMODEL + col] = w * (acc[rs][j][r] + be);
            }
        }
    }
}

// ---------------------------------------------------------------- combine

__global__ void combine_kernel(float* __restrict__ out,
                               const float* __restrict__ rk0,
                               const float* __restrict__ rk1) {
    int i = blockIdx.x * blockDim.x + threadIdx.x;
    if (i < S_TOK * DMODEL) out[i] = out[i] + rk0[i] + rk1[i];
}

// ---------------------------------------------------------------- launcher

extern "C" void kernel_launch(void* const* d_in, const int* in_sizes, int n_in,
                              void* d_out, int out_size, void* d_ws, size_t ws_size,
                              hipStream_t stream) {
    const float* x               = (const float*)d_in[0];
    const float* ln_scale        = (const float*)d_in[1];
    const float* ln_bias         = (const float*)d_in[2];
    const float* w_shared_in     = (const float*)d_in[3];
    const float* w_shared_out    = (const float*)d_in[4];
    const float* b_shared_out    = (const float*)d_in[5];
    const float* w_expert_in     = (const float*)d_in[6];
    const float* expert_out_w    = (const float*)d_in[7];
    const float* expert_out_b    = (const float*)d_in[8];
    const float* w_group_gate    = (const float*)d_in[9];
    const float* w_expert_gate   = (const float*)d_in[10];
    const float* group_bias      = (const float*)d_in[11];
    const float* expert_bias     = (const float*)d_in[12];
    float* out = (float*)d_out;

    char* ws = (char*)d_ws;
    size_t off = 0;
    auto alloc = [&](size_t bytes) -> void* {
        void* p = ws + off;
        off = (off + bytes + 255) & ~(size_t)255;
        return p;
    };
    bf16_t* flatbf = (bf16_t*)alloc((size_t)S_TOK * DMODEL * 2);
    bf16_t* hs     = (bf16_t*)alloc((size_t)S_TOK * HDIM * 2);
    bf16_t* hx     = (bf16_t*)alloc((size_t)(S_TOK + 16) * HDIM * 2);
    bf16_t* wsi_t  = (bf16_t*)alloc((size_t)2 * HDIM * DMODEL * 2);
    bf16_t* wso_t  = (bf16_t*)alloc((size_t)DMODEL * HDIM * 2);
    bf16_t* wei_t  = (bf16_t*)alloc((size_t)2 * HDIM * DMODEL * 2);
    bf16_t* weo_bf = (bf16_t*)alloc((size_t)NEXP * DMODEL * HDIM * 2);
    float*  rk0    = (float*)alloc((size_t)(S_TOK + 16) * DMODEL * 4);
    float*  rk1    = (float*)alloc((size_t)(S_TOK + 16) * DMODEL * 4);
    int*    stok   = (int*)alloc((size_t)NEXP * CAP * 4);
    float*  swt    = (float*)alloc((size_t)NEXP * CAP * 4);
    int*    sslot  = (int*)alloc((size_t)NEXP * CAP * 4);
    int*    cursor = (int*)alloc(NEXP * 4);

    // 0) init sorted arrays / cursors / dummy row
    init_kernel<<<64, 256, 0, stream>>>(stok, swt, sslot, cursor, hx);

    // 1) layernorm -> bf16 flat
    ln_kernel<<<S_TOK / 8, 256, 0, stream>>>(x, ln_scale, ln_bias, flatbf);

    // 2) weight prep
    {
        dim3 b(32, 8);
        transpose_cvt_kernel<<<dim3(2 * HDIM / 32, DMODEL / 32), b, 0, stream>>>(
            w_shared_in, wsi_t, DMODEL, 2 * HDIM);
        transpose_cvt_kernel<<<dim3(DMODEL / 32, HDIM / 32), b, 0, stream>>>(
            w_shared_out, wso_t, HDIM, DMODEL);
        transpose_cvt_kernel<<<dim3(2 * HDIM / 32, DMODEL / 32), b, 0, stream>>>(
            w_expert_in, wei_t, DMODEL, 2 * HDIM);
        int n = NEXP * DMODEL * HDIM;
        cvt_kernel<<<(n + 255) / 256, 256, 0, stream>>>(expert_out_w, weo_bf, n);
    }

    // 3) routing (fills stok/swt/sslot, cursor[e] = live count)
    route_kernel<<<S_TOK / 8, 256, 0, stream>>>(
        flatbf, w_group_gate, w_expert_gate, group_bias, expert_bias,
        stok, swt, sslot, cursor);

    // 4) shared + expert in-proj with fused SwiGLU (32-row wave tiles)
    gemm_swiglu_kernel<<<dim3(S_TOK / 256, HDIM / 64), 256, 0, stream>>>(
        flatbf, wsi_t, hs);
    gemm_swiglu_kernel<<<dim3(S_TOK / 256, HDIM / 64), 256, 0, stream>>>(
        flatbf, wei_t, hx);

    // 5) shared out-proj -> d_out (fp32, + bias)
    gemm_bias_kernel<<<dim3(S_TOK / 256, DMODEL / 64), 256, 0, stream>>>(
        hs, wso_t, b_shared_out, out);

    // 6) grouped expert out-proj -> rk0/rk1 (weighted, + expert bias)
    gemm_expert_kernel<<<dim3(CAP / 32, DMODEL / 64, NEXP), 32, 0, stream>>>(
        hx, weo_bf, expert_out_b, stok, swt, sslot, cursor, rk0, rk1);

    // 7) combine
    combine_kernel<<<(S_TOK * DMODEL + 255) / 256, 256, 0, stream>>>(out, rk0, rk1);
}
